// Flashback_87737591923386
// MI455X (gfx1250) — matrix operand order repaired and checked by reading, not compiled, not run on hardware
//
#include <hip/hip_runtime.h>
#include <math.h>

#define S_ 128
#define U_ 128
#define H_ 512
#define V_ 8192

typedef unsigned short u16;
typedef __attribute__((ext_vector_type(8)))  __bf16 bf8;
typedef __attribute__((ext_vector_type(16))) __bf16 bf16x16;
typedef __attribute__((ext_vector_type(8)))  float  f32x8;

// ---------- helpers ----------

static __device__ __forceinline__ u16 f2bf(float f) {
  union { float f; unsigned u; } v; v.f = f;
  unsigned r = v.u + 0x7FFFu + ((v.u >> 16) & 1u);   // round-to-nearest-even
  return (u16)(r >> 16);
}

// A fragment: 16x32 bf16, row m = m0 + (lane&15).
// Contiguous K order (matches B): lanes 0-15 hold K = 0..15;
// lanes 16-31 hold K = 16..31.
static __device__ __forceinline__ bf16x16 load_a_frag(const u16* __restrict__ base,
                                                      int lda, int m0, int kbase, int lane) {
  const int r = lane & 15, hb = lane >> 4;
  const u16* p = base + (size_t)(m0 + r) * lda + kbase + hb * 16;
  bf8 lo = *reinterpret_cast<const bf8*>(p);
  bf8 hi = *reinterpret_cast<const bf8*>(p + 8);
  bf16x16 f;
#pragma unroll
  for (int i = 0; i < 8; ++i) { f[i] = lo[i]; f[i + 8] = hi[i]; }
  return f;
}

// B fragment: 32x16 bf16, column n = n0 + (lane&15); lanes 0-15 K=0..15,
// lanes 16-31 K=16..31. Columns must be K-contiguous with leading dim ldb.
static __device__ __forceinline__ bf16x16 load_b_frag(const u16* __restrict__ base,
                                                      int ldb, int n0, int kbase, int lane) {
  const u16* p = base + (size_t)(n0 + (lane & 15)) * ldb + kbase + (lane >> 4) * 16;
  bf8 lo = *reinterpret_cast<const bf8*>(p);
  bf8 hi = *reinterpret_cast<const bf8*>(p + 8);
  bf16x16 f;
#pragma unroll
  for (int i = 0; i < 8; ++i) { f[i] = lo[i]; f[i + 8] = hi[i]; }
  return f;
}

#define WMMA_BF16(a, b, c) \
  __builtin_amdgcn_wmma_f32_16x16x32_bf16(false, (a), false, (b), (short)0, (c), false, false)

// ---------- prep kernels ----------

__global__ void k_cvt_bf16(const float* __restrict__ in, u16* __restrict__ out, int n) {
  int i = blockIdx.x * blockDim.x + threadIdx.x;
  if (i < n) out[i] = f2bf(in[i]);
}

__global__ void k_bias_sum(const float* __restrict__ bih, const float* __restrict__ bhh,
                           float* __restrict__ bsum) {
  int i = blockIdx.x * blockDim.x + threadIdx.x;
  if (i < H_) bsum[i] = bih[i] + bhh[i];
}

__global__ void k_gather_x(const int* __restrict__ x, const float* __restrict__ emb,
                           u16* __restrict__ Xb) {
  int i = blockIdx.x * blockDim.x + threadIdx.x;       // over S*U*H
  if (i >= S_ * U_ * H_) return;
  int h = i & (H_ - 1);
  int su = i >> 9;                                      // H_ = 512
  Xb[i] = f2bf(emb[(size_t)x[su] * H_ + h]);
}

__global__ void k_fill_pu(const int* __restrict__ act, const float* __restrict__ uemb,
                          u16* __restrict__ Cb) {
  int i = blockIdx.x * blockDim.x + threadIdx.x;       // over S*U*H
  if (i >= S_ * U_ * H_) return;
  int h = i & (H_ - 1);
  int su = i >> 9;
  int u = su & (U_ - 1);
  Cb[(size_t)su * (2 * H_) + H_ + h] = f2bf(uemb[(size_t)act[u] * H_ + h]);
}

// ---------- A = Xb @ Wih^T + (b_ih + b_hh)  [16384 x 512] ----------

__global__ void k_gemm_A(const u16* __restrict__ Xb, const u16* __restrict__ Wb,
                         const float* __restrict__ bias, float* __restrict__ A) {
  const int lane = threadIdx.x & 31;
  const int wave = threadIdx.x >> 5;
  const int m0 = (blockIdx.x * 4 + wave) * 16;
  const int n0 = blockIdx.y * 16;
  f32x8 acc = {};
#pragma unroll
  for (int k = 0; k < H_; k += 32) {
    bf16x16 a = load_a_frag(Xb, H_, m0, k, lane);
    bf16x16 b = load_b_frag(Wb, H_, n0, k, lane);
    acc = WMMA_BF16(a, b, acc);
  }
  const int r = lane & 15, hb = lane >> 4;
  const float bn = bias[n0 + r];
#pragma unroll
  for (int v = 0; v < 8; ++v)
    A[(size_t)(m0 + v + 8 * hb) * H_ + n0 + r] = acc[v] + bn;
}

// ---------- one RNN step: h_new = tanh(A_t + h_prev @ Whh^T) ----------
// Writes h_new twice: [U][H] ping-pong slot (A-fragment layout for next step)
// and transposed [U][H][S] (K-contiguous B layout for the weighted-sum GEMM).

__global__ void k_rnn_step(const u16* __restrict__ Hprev, const u16* __restrict__ Whh,
                           const float* __restrict__ At, u16* __restrict__ Hnext,
                           u16* __restrict__ OutT, int tstep,
                           float* __restrict__ HoutF32) {
  const int lane = threadIdx.x & 31;
  const int wave = threadIdx.x >> 5;
  const int m0 = (blockIdx.x * 4 + wave) * 16;          // user rows (U=128)
  const int n0 = blockIdx.y * 16;                       // hidden cols (H=512)
  f32x8 acc = {};
#pragma unroll
  for (int k = 0; k < H_; k += 32) {
    bf16x16 a = load_a_frag(Hprev, H_, m0, k, lane);
    bf16x16 b = load_b_frag(Whh, H_, n0, k, lane);
    acc = WMMA_BF16(a, b, acc);
  }
  const int r = lane & 15, hb = lane >> 4;
#pragma unroll
  for (int v = 0; v < 8; ++v) {
    const int m = m0 + v + 8 * hb;                      // user
    const int n = n0 + r;                               // hidden
    const size_t idx = (size_t)m * H_ + n;
    float hval = tanhf(acc[v] + At[idx]);
    u16 hb16 = f2bf(hval);
    Hnext[idx] = hb16;
    OutT[((size_t)m * H_ + n) * S_ + tstep] = hb16;     // transposed copy
    if (HoutF32) HoutF32[idx] = hval;                   // final hidden state (f32)
  }
}

// ---------- flashback weights w[u][i][j] (bf16) + sum_w[i][u] (f32) ----------

__global__ void k_flash_w(const float* __restrict__ t, const float* __restrict__ s,
                          u16* __restrict__ Wmat, float* __restrict__ sumw) {
  int idx = blockIdx.x * blockDim.x + threadIdx.x;      // i*U + u
  if (idx >= S_ * U_) return;
  const int i = idx >> 7, u = idx & (U_ - 1);           // U_ = 128
  const float ti  = t[i * U_ + u];
  const float si0 = s[((size_t)i * U_ + u) * 2 + 0];
  const float si1 = s[((size_t)i * U_ + u) * 2 + 1];
  const float CYC = (float)(2.0 * M_PI / 86400.0);
  const float INV_DAY = 1.0f / 86400.0f;
  float sum = 0.0f;
  u16* wrow = Wmat + ((size_t)u * S_ + i) * S_;
  for (int j = 0; j < S_; ++j) {
    float w = 0.0f;
    if (j <= i) {
      float dt = ti - t[j * U_ + u];
      float d0 = si0 - s[((size_t)j * U_ + u) * 2 + 0];
      float d1 = si1 - s[((size_t)j * U_ + u) * 2 + 1];
      float ds = sqrtf(d0 * d0 + d1 * d1);
      float a = (__cosf(dt * CYC) + 1.0f) * 0.5f * __expf(-dt * INV_DAY * 0.1f);
      float b = __expf(-ds * 100.0f);
      w = a * b + 1e-10f;
    }
    sum += w;
    wrow[j] = f2bf(w);
  }
  sumw[idx] = sum;
}

// ---------- per-user weighted sum: out_w[i,u,:] = (Wmat_u @ Out_u) / sum_w ----------
// B comes from the transposed RNN outputs OutT[u][h][t]: K-contiguous columns.

__global__ void k_outw(const u16* __restrict__ Wmat, const u16* __restrict__ OutT,
                       const float* __restrict__ sumw, u16* __restrict__ Cb) {
  const int lane = threadIdx.x & 31;
  const int wave = threadIdx.x >> 5;
  const int u = blockIdx.z;
  const int m0 = (blockIdx.x * 4 + wave) * 16;          // time rows (S=128)
  const int n0 = blockIdx.y * 16;                       // hidden cols
  const u16* Au = Wmat + (size_t)u * S_ * S_;
  const u16* Bu = OutT + (size_t)u * H_ * S_;           // B[k][n] = OutT[u][n][k]
  f32x8 acc = {};
#pragma unroll
  for (int k = 0; k < S_; k += 32) {
    bf16x16 a = load_a_frag(Au, S_, m0, k, lane);
    bf16x16 b = load_b_frag(Bu, S_, n0, k, lane);
    acc = WMMA_BF16(a, b, acc);
  }
  const int r = lane & 15, hb = lane >> 4;
#pragma unroll
  for (int v = 0; v < 8; ++v) {
    const int i = m0 + v + 8 * hb;
    float ow = acc[v] / sumw[i * U_ + u];
    Cb[((size_t)i * U_ + u) * (2 * H_) + n0 + r] = f2bf(ow);
  }
}

// ---------- final FC: Y[16384 x 8192] = Cb @ fcW^T + fc_b ----------
// Block = 4 waves = 128x64 block tile; B k-chunks staged in LDS (double
// buffered, padded stride), each wave computes a 32x64 register tile.

#define BSTRIDE 40   // LDS row stride in elements (80B, 16B-aligned, bank-spread)

__global__ void k_fc(const u16* __restrict__ Cb, const u16* __restrict__ Wb,
                     const float* __restrict__ fcb, float* __restrict__ Y) {
  __shared__ u16 Bs[2][64 * BSTRIDE];
  const int tid  = threadIdx.x;
  const int lane = tid & 31;
  const int wave = tid >> 5;
  const int m0 = blockIdx.y * 128 + wave * 32;
  const int n0 = blockIdx.x * 64;

  // staging pattern: thread -> (row n = tid>>1 of 64, half = tid&1), 32B each
  const int srow = tid >> 1, shalf = tid & 1;
  const u16* gB = Wb + (size_t)(n0 + srow) * (2 * H_) + shalf * 16;

  // preload k=0 chunk
  bf8 r0 = *reinterpret_cast<const bf8*>(gB + 0);
  bf8 r1 = *reinterpret_cast<const bf8*>(gB + 8);

  f32x8 acc[8] = {};                                    // [mt(2)][nt(4)]

  int cur = 0;
  for (int k = 0; k < 2 * H_; k += 32) {
    // commit staged chunk to LDS[cur]
    u16* dst = &Bs[cur][srow * BSTRIDE + shalf * 16];
    *reinterpret_cast<bf8*>(dst + 0) = r0;
    *reinterpret_cast<bf8*>(dst + 8) = r1;
    __syncthreads();
    // prefetch next chunk from global (overlaps with WMMA below)
    if (k + 32 < 2 * H_) {
      r0 = *reinterpret_cast<const bf8*>(gB + (k + 32));
      r1 = *reinterpret_cast<const bf8*>(gB + (k + 32) + 8);
    }
    // A fragments from global (L2-resident)
    bf16x16 a0 = load_a_frag(Cb, 2 * H_, m0,      k, lane);
    bf16x16 a1 = load_a_frag(Cb, 2 * H_, m0 + 16, k, lane);
#pragma unroll
    for (int t = 0; t < 4; ++t) {
      const u16* p = &Bs[cur][(t * 16 + (lane & 15)) * BSTRIDE + (lane >> 4) * 16];
      bf8 lo = *reinterpret_cast<const bf8*>(p);
      bf8 hi = *reinterpret_cast<const bf8*>(p + 8);
      bf16x16 b;
#pragma unroll
      for (int i = 0; i < 8; ++i) { b[i] = lo[i]; b[i + 8] = hi[i]; }
      acc[t]     = WMMA_BF16(a0, b, acc[t]);
      acc[4 + t] = WMMA_BF16(a1, b, acc[4 + t]);
    }
    cur ^= 1;
  }

  const int r = lane & 15, hb = lane >> 4;
#pragma unroll
  for (int t = 0; t < 4; ++t) {
    const int n = n0 + t * 16 + r;
    const float bn = fcb[n];
#pragma unroll
    for (int mt = 0; mt < 2; ++mt) {
#pragma unroll
      for (int v = 0; v < 8; ++v)
        Y[(size_t)(m0 + mt * 16 + v + 8 * hb) * V_ + n] = acc[mt * 4 + t][v] + bn;
    }
  }
}

// ---------- host ----------

extern "C" void kernel_launch(void* const* d_in, const int* in_sizes, int n_in,
                              void* d_out, int out_size, void* d_ws, size_t ws_size,
                              hipStream_t stream) {
  const int*   x        = (const int*)  d_in[0];   // [S,U]
  const float* t_in     = (const float*)d_in[1];   // [S,U]
  const float* s_in     = (const float*)d_in[2];   // [S,U,2]
  //            y_t, y_s (d_in[3], d_in[4]) unused
  const float* h0       = (const float*)d_in[5];   // [1,U,H]
  const int*   act      = (const int*)  d_in[6];   // [U]
  const float* enc_emb  = (const float*)d_in[7];   // [V,H]
  const float* user_emb = (const float*)d_in[8];   // [UC,H]
  const float* W_ih     = (const float*)d_in[9];   // [H,H]
  const float* W_hh     = (const float*)d_in[10];  // [H,H]
  const float* b_ih     = (const float*)d_in[11];  // [H]
  const float* b_hh     = (const float*)d_in[12];  // [H]
  const float* fc_W     = (const float*)d_in[13];  // [V,2H]
  const float* fc_b     = (const float*)d_in[14];  // [V]

  float* Y = (float*)d_out;                                 // [S,U,V]
  float* HoutF32 = Y + (size_t)S_ * U_ * V_;                // [U,H] tail

  char* ws = (char*)d_ws;
  size_t off = 0;
  auto carve = [&](size_t bytes) -> char* {
    char* p = ws + off;
    off = (off + bytes + 255) & ~(size_t)255;
    return p;
  };
  u16*   Xb    = (u16*)  carve((size_t)S_ * U_ * H_ * 2);         // 16 MB
  u16*   WihB  = (u16*)  carve((size_t)H_ * H_ * 2);
  u16*   WhhB  = (u16*)  carve((size_t)H_ * H_ * 2);
  u16*   fcWB  = (u16*)  carve((size_t)V_ * 2 * H_ * 2);          // 16 MB
  float* bsum  = (float*)carve((size_t)H_ * 4);
  float* Apre  = (float*)carve((size_t)S_ * U_ * H_ * 4);         // 32 MB
  u16*   Hping = (u16*)  carve((size_t)2 * U_ * H_ * 2);          // ping-pong h
  u16*   OutT  = (u16*)  carve((size_t)U_ * H_ * S_ * 2);         // 16 MB transposed
  u16*   Wmat  = (u16*)  carve((size_t)U_ * S_ * S_ * 2);         // 4 MB
  float* sumw  = (float*)carve((size_t)S_ * U_ * 4);
  u16*   Cb    = (u16*)  carve((size_t)S_ * U_ * 2 * H_ * 2);     // 32 MB
  (void)ws_size; (void)n_in; (void)in_sizes; (void)out_size;

  const int TB = 256;
  // --- prep / conversions ---
  k_cvt_bf16<<<(H_ * H_ + TB - 1) / TB, TB, 0, stream>>>(W_ih, WihB, H_ * H_);
  k_cvt_bf16<<<(H_ * H_ + TB - 1) / TB, TB, 0, stream>>>(W_hh, WhhB, H_ * H_);
  k_cvt_bf16<<<(V_ * 2 * H_ + TB - 1) / TB, TB, 0, stream>>>(fc_W, fcWB, V_ * 2 * H_);
  k_cvt_bf16<<<(U_ * H_ + TB - 1) / TB, TB, 0, stream>>>(h0, Hping /* slot 0 */, U_ * H_);
  k_bias_sum<<<(H_ + TB - 1) / TB, TB, 0, stream>>>(b_ih, b_hh, bsum);
  k_gather_x<<<(S_ * U_ * H_ + TB - 1) / TB, TB, 0, stream>>>(x, enc_emb, Xb);
  k_fill_pu<<<(S_ * U_ * H_ + TB - 1) / TB, TB, 0, stream>>>(act, user_emb, Cb);

  // --- A = Xb @ Wih^T + bias (parallel over all timesteps) ---
  k_gemm_A<<<dim3((S_ * U_ / 16) / 4, H_ / 16), 128, 0, stream>>>(Xb, WihB, bsum, Apre);

  // --- sequential RNN: 128 small WMMA steps (ping-pong hidden state) ---
  for (int tstep = 0; tstep < S_; ++tstep) {
    const u16*  hp = Hping + (size_t)(tstep & 1) * U_ * H_;
    u16*        hn = Hping + (size_t)((tstep + 1) & 1) * U_ * H_;
    const float* At = Apre + (size_t)tstep * U_ * H_;
    float* hout = (tstep == S_ - 1) ? HoutF32 : nullptr;
    k_rnn_step<<<dim3((U_ / 16) / 4, H_ / 16), 128, 0, stream>>>(
        hp, WhhB, At, hn, OutT, tstep, hout);
  }

  // --- flashback weights ---
  k_flash_w<<<(S_ * U_ + TB - 1) / TB, TB, 0, stream>>>(t_in, s_in, Wmat, sumw);

  // --- out_w = (w @ out) / sum_w, per user; writes left half of Cb ---
  k_outw<<<dim3((S_ / 16) / 4, H_ / 16, U_), 128, 0, stream>>>(Wmat, OutT, sumw, Cb);

  // --- final FC GEMM (dominant): LDS-staged B, 128x64 block tiles ---
  k_fc<<<dim3(V_ / 64, (S_ * U_) / 128), 128, 0, stream>>>(Cb, fcWB, fc_b, Y);
}
